// MS_37031208026268
// MI455X (gfx1250) — matrix operand order repaired and checked by reading, not compiled, not run on hardware
//
#include <hip/hip_runtime.h>
#include <hip/hip_bf16.h>

// Mean-shift iterations, fused flash-attention style (K never materialized).
//   S = (s*x)^T (s*x)  -> v_wmma_f32_16x16x32_bf16, K-depth 32 == EMBED_DIM,
//                         with s = sqrt(bw*log2e) folded into xT so the WMMA
//                         output is directly the exp2 argument (no v_mul).
//   P = exp2(S)        -> v_exp_f32, P stays in registers
//   out^T += P^T-as-A * x^T   (Gram C-layout == A-operand layout, no shuffles)
// deg_j via in-lane sums + shfl_xor(16). Software-pipelined operand loads +
// global_prefetch two chunks ahead.

#define NN 8192
#define DD 32
#define WPB 4   // waves per block (wave32)

typedef __attribute__((ext_vector_type(16))) __bf16      v16bf;
typedef __attribute__((ext_vector_type(8)))  float       v8f;
typedef __attribute__((ext_vector_type(4)))  unsigned int u32x4;

union BF16x16 {
    v16bf v;
    u32x4 q[2];
};

__global__ __launch_bounds__(256) void ms_convert(const float* __restrict__ x,
                                                  __bf16* __restrict__ xbf,
                                                  __bf16* __restrict__ xTbf) {
    int t = blockIdx.x * blockDim.x + threadIdx.x;   // 0 .. 32*8192-1
    int n = t & (NN - 1);
    int d = t >> 13;
    float f = x[t];
    xbf[t] = (__bf16)f;                              // [32][8192], unscaled values
    // Gram operands pre-scaled by sqrt(bw * log2(e)) so WMMA emits exp2 args.
    xTbf[n * DD + d] = (__bf16)(f * 0.84932180f);    // [8192][32], 64B rows
}

__global__ __launch_bounds__(32 * WPB) void ms_iter(const float* __restrict__ xin,
                                                    const __bf16* __restrict__ xbf,   // [32][8192]
                                                    const __bf16* __restrict__ xTbf,  // [8192][32] scaled
                                                    float* __restrict__ xout) {
    const int lane = threadIdx.x & 31;
    const int wave = threadIdx.x >> 5;
    const int m    = lane & 15;       // lane within half-wave
    const int hi   = lane >> 4;       // which half-wave
    const int j_base = (blockIdx.x * WPB + wave) * 16;

    // --- Gram B operand (loop invariant): column j = j_base+m, d = hi*8 + 0..7, 16 + hi*8 + 0..7
    BF16x16 bj;
    {
        const u32x4* p = (const u32x4*)(xTbf + (j_base + m) * DD + hi * 8);
        bj.q[0] = p[0];
        bj.q[1] = p[2];
    }

    auto loadA = [&](int i0, BF16x16& a0, BF16x16& a1) {
        const __bf16* r0 = xTbf + (i0 + m) * DD;
        a0.q[0] = *(const u32x4*)(r0 + hi * 8);          // d = hi*8 + 0..7
        a0.q[1] = *(const u32x4*)(r0 + 16 + hi * 8);     // d = 16 + hi*8 + 0..7
        const __bf16* r1 = xTbf + (i0 + 16 + m) * DD;
        a1.q[0] = *(const u32x4*)(r1 + hi * 8);
        a1.q[1] = *(const u32x4*)(r1 + 16 + hi * 8);
    };
    auto loadB2 = [&](int i0, BF16x16& lo, BF16x16& hh) {
        const __bf16* p0 = xbf + m * NN + i0 + hi * 8;
        lo.q[0] = *(const u32x4*)(p0);
        lo.q[1] = *(const u32x4*)(p0 + 16);
        const __bf16* p1 = xbf + (16 + m) * NN + i0 + hi * 8;
        hh.q[0] = *(const u32x4*)(p1);
        hh.q[1] = *(const u32x4*)(p1 + 16);
    };

    v8f accT0 = {};     // out^T tile, d' = 0..15
    v8f accT1 = {};     // out^T tile, d' = 16..31
    float deg = 0.0f;

    // software pipeline: current-chunk operands pre-loaded
    BF16x16 a0c, a1c, b2lc, b2hc;
    loadA(0, a0c, a1c);
    loadB2(0, b2lc, b2hc);

    for (int i0 = 0; i0 < NN; i0 += 32) {
        // prefetch two chunks ahead (speculative; dropped if OOB)
        __builtin_prefetch(xTbf + (i0 + 64 + m) * DD, 0, 3);
        __builtin_prefetch(xbf + m * NN + i0 + 64, 0, 3);
        __builtin_prefetch(xbf + (16 + m) * NN + i0 + 64, 0, 3);

        // --- Gram: S tiles (16i x 16j), full K-depth 32 in one WMMA each
        v8f z = {};
        v8f c0 = __builtin_amdgcn_wmma_f32_16x16x32_bf16(false, a0c.v, false, bj.v,
                                                         (short)0, z, false, false);
        v8f c1 = __builtin_amdgcn_wmma_f32_16x16x32_bf16(false, a1c.v, false, bj.v,
                                                         (short)0, z, false, false);

        // --- issue next chunk's loads while WMMAs/exps run (fills hazard slots)
        const int in = (i0 + 32 < NN) ? (i0 + 32) : i0;   // clamp last iter
        BF16x16 a0n, a1n, b2ln, b2hn;
        loadA(in, a0n, a1n);
        loadB2(in, b2ln, b2hn);

        // --- exp2(S): stays in-lane; C-layout of c0/c1 is exactly the
        //     A-operand layout of the transposed second GEMM.
        BF16x16 a2;
#pragma unroll
        for (int r = 0; r < 8; ++r) {
            float p0 = __builtin_amdgcn_exp2f(c0[r]);   // i = i0 + r + hi*8
            float p1 = __builtin_amdgcn_exp2f(c1[r]);   // i = i0 + 16 + r + hi*8
            deg += p0 + p1;
            a2.v[r]     = (__bf16)p0;
            a2.v[r + 8] = (__bf16)p1;
        }

        // --- out^T accumulation: A2 = P^T (in-register), B = x^T chunk
        accT0 = __builtin_amdgcn_wmma_f32_16x16x32_bf16(false, a2.v, false, b2lc.v,
                                                        (short)0, accT0, false, false);
        accT1 = __builtin_amdgcn_wmma_f32_16x16x32_bf16(false, a2.v, false, b2hc.v,
                                                        (short)0, accT1, false, false);

        a0c = a0n; a1c = a1n; b2lc = b2ln; b2hc = b2hn;
    }

    // --- deg_j: fold the two half-waves, then broadcast per output row
    float degF = deg + __shfl_xor(deg, 16, 32);   // lane L: deg for j = j_base + (L&15)
    float dg[8];
#pragma unroll
    for (int r = 0; r < 8; ++r)
        dg[r] = __shfl(degF, hi * 8 + r, 32);     // deg for j = j_base + hi*8 + r

    // --- epilogue: lane L owns d = h*16 + m, j = j_base + hi*8 + 0..7 (contiguous)
    const int j0 = j_base + hi * 8;
#pragma unroll
    for (int h = 0; h < 2; ++h) {
        v8f acc = h ? accT1 : accT0;
        const int d = h * 16 + m;
        const float* xr = xin + d * NN + j0;
        float4 xi0 = *(const float4*)(xr);
        float4 xi1 = *(const float4*)(xr + 4);
        float4 o0, o1;
        o0.x = 0.5f * acc[0] / dg[0] + 0.5f * xi0.x;
        o0.y = 0.5f * acc[1] / dg[1] + 0.5f * xi0.y;
        o0.z = 0.5f * acc[2] / dg[2] + 0.5f * xi0.z;
        o0.w = 0.5f * acc[3] / dg[3] + 0.5f * xi0.w;
        o1.x = 0.5f * acc[4] / dg[4] + 0.5f * xi1.x;
        o1.y = 0.5f * acc[5] / dg[5] + 0.5f * xi1.y;
        o1.z = 0.5f * acc[6] / dg[6] + 0.5f * xi1.z;
        o1.w = 0.5f * acc[7] / dg[7] + 0.5f * xi1.w;
        float* xo = xout + d * NN + j0;
        *(float4*)(xo)     = o0;
        *(float4*)(xo + 4) = o1;
    }
}

extern "C" void kernel_launch(void* const* d_in, const int* in_sizes, int n_in,
                              void* d_out, int out_size, void* d_ws, size_t ws_size,
                              hipStream_t stream) {
    (void)in_sizes; (void)n_in; (void)out_size; (void)ws_size;

    const float* x0 = (const float*)d_in[0];      // [32][8192] fp32
    float* outbuf = (float*)d_out;                // [32][8192] fp32

    // workspace: 1MB fp32 ping buffer + 512KB x_bf16 + 512KB xT_bf16 = 2MB
    float*  xA   = (float*)d_ws;
    __bf16* xbf  = (__bf16*)((char*)d_ws + (size_t)DD * NN * sizeof(float));
    __bf16* xTbf = xbf + (size_t)DD * NN;

    const float* cur = x0;
    for (int it = 0; it < 5; ++it) {
        ms_convert<<<(DD * NN) / 256, 256, 0, stream>>>(cur, xbf, xTbf);
        float* out = (it & 1) ? xA : outbuf;      // iters 0,2,4 -> d_out; 1,3 -> ws
        ms_iter<<<NN / (16 * WPB), 32 * WPB, 0, stream>>>(cur, xbf, xTbf, out);
        cur = out;
    }
    // it==4 wrote d_out
}
